// MaskedMHA_49675591746216
// MI455X (gfx1250) — matrix operand order repaired and checked by reading, not compiled, not run on hardware
//
#include <hip/hip_runtime.h>
#include <hip/hip_bf16.h>
#include <stdint.h>

// Problem constants (reference: B=32, L=1024, H=8, D=64, M=512)
#define BB 32
#define LL 1024
#define HH 8
#define DD 64
#define MM 512
#define BL (BB * LL)   // 32768

typedef __bf16 bf16_t;
typedef __attribute__((ext_vector_type(16))) __bf16 v16bf;
typedef __attribute__((ext_vector_type(8)))  float  v8f;
typedef __attribute__((ext_vector_type(4)))  uint32_t u32x4;
typedef __attribute__((ext_vector_type(8)))  uint32_t u32x8;

union Frag16 { v16bf v; uint4 q[2]; };

__device__ __forceinline__ v8f wmma_bf16(v16bf a, v16bf b, v8f c) {
  return __builtin_amdgcn_wmma_f32_16x16x32_bf16(
      /*neg_a=*/false, a, /*neg_b=*/false, b,
      /*c_mod=*/(short)0, c, /*reuse_a=*/false, /*reuse_b=*/false);
}

// Async global->LDS copy of 16 bytes per lane (ASYNCcnt path).
// LDS offset = low 32 bits of the generic pointer (LDS aperture truncation).
__device__ __forceinline__ void async_copy_b128(void* lds_dst, const void* g_src) {
  uint32_t lds_off = (uint32_t)(uintptr_t)lds_dst;
  asm volatile("global_load_async_to_lds_b128 %0, %1, off"
               :: "v"(lds_off), "v"(g_src) : "memory");
}
__device__ __forceinline__ void wait_async() {
  asm volatile("s_wait_asynccnt 0x0" ::: "memory");
}

// ---------------------------------------------------------------------------
// f32 -> bf16 conversion (8 contiguous elements per thread)
// ---------------------------------------------------------------------------
__global__ __launch_bounds__(256) void cvt_f32_to_bf16(
    const float* __restrict__ src, bf16_t* __restrict__ dst, int n8) {
  int i = blockIdx.x * blockDim.x + threadIdx.x;
  if (i >= n8) return;
  const float4* s = (const float4*)src;
  float4 a = s[2 * i + 0];
  float4 b = s[2 * i + 1];
  bf16_t tmp[8];
  tmp[0] = (bf16_t)a.x; tmp[1] = (bf16_t)a.y; tmp[2] = (bf16_t)a.z; tmp[3] = (bf16_t)a.w;
  tmp[4] = (bf16_t)b.x; tmp[5] = (bf16_t)b.y; tmp[6] = (bf16_t)b.z; tmp[7] = (bf16_t)b.w;
  ((uint4*)dst)[i] = *(const uint4*)tmp;
}

// ---------------------------------------------------------------------------
// Transposing f32 -> bf16 weight conversion: in (z, K, N) f32 -> out (z, N, K) bf16.
// Makes GEMM B tiles contiguous so they can be staged with async loads.
// ---------------------------------------------------------------------------
__global__ __launch_bounds__(256) void transpose_cvt_bf16(
    const float* __restrict__ in, bf16_t* __restrict__ out, int K, int N) {
  const int z = blockIdx.z;
  in  += (size_t)z * K * N;
  out += (size_t)z * K * N;
  int idx = blockIdx.x * 256 + threadIdx.x;  // over N*(K/8)
  int n  = idx / (K / 8);
  int k8 = (idx % (K / 8)) * 8;
  if (n >= N) return;
  bf16_t tmp[8];
#pragma unroll
  for (int j = 0; j < 8; ++j) tmp[j] = (bf16_t)in[(size_t)(k8 + j) * N + n];
  *(uint4*)(out + (size_t)n * K + k8) = *(const uint4*)tmp;
}

// ---------------------------------------------------------------------------
// Generic bf16 WMMA GEMM with f32 bias: C[row,col] = A(row,:) . Bt(col,:) + bias[col]
//   A:  (BL x K) bf16 row-major (shared across z)
//   Bt: (N x K)  bf16 row-major (TRANSPOSED weights), offset z*Bstride
// Workgroup: 256 threads (8 waves); tile 256 rows x 64 cols; wave -> 32 x 64.
// Staging uses global_load_async_to_lds_b128 (ASYNCcnt).
// ---------------------------------------------------------------------------
template <bool OUT_BF16>
__global__ __launch_bounds__(256) void gemm_bias_wmma(
    const bf16_t* __restrict__ A, const bf16_t* __restrict__ Bt,
    const float* __restrict__ bias, float* __restrict__ outF,
    bf16_t* __restrict__ outH, int K, int N,
    long Bstride, int biasStride, long outStride) {
  const int z = blockIdx.z;
  const bf16_t* Bmat = Bt + (size_t)z * Bstride;
  const float*  bz   = bias + (size_t)z * biasStride;
  const int rowBase = blockIdx.x * 256;
  const int cb      = blockIdx.y * 64;

  __shared__ bf16_t Al[256][40];  // 256x32 A tile, padded rows (80B, 16B aligned)
  __shared__ bf16_t Bl[64][40];   // 64 cols x 32 k (from transposed weights)

  const int tid  = threadIdx.x;
  const int wave = tid >> 5;
  const int lane = tid & 31;
  const int half = lane >> 4;
  const int ln   = lane & 15;

  v8f acc[2][4];
#pragma unroll
  for (int r = 0; r < 2; ++r)
#pragma unroll
    for (int t = 0; t < 4; ++t)
      acc[r][t] = (v8f){0.f, 0.f, 0.f, 0.f, 0.f, 0.f, 0.f, 0.f};

  for (int kb = 0; kb < K; kb += 32) {
    __syncthreads();
    // Async-stage A tile: 256 rows x 32 k = 1024 x b128 chunks.
#pragma unroll
    for (int c = tid; c < 1024; c += 256) {
      int r  = c >> 2;
      int c8 = (c & 3) * 8;
      async_copy_b128(&Al[r][c8], A + (size_t)(rowBase + r) * K + kb + c8);
    }
    // Async-stage B tile: Bl[n][kk] = Bt[cb+n][kb+kk]; 256 x b128 chunks.
    {
      int n  = tid >> 2;
      int c8 = (tid & 3) * 8;
      async_copy_b128(&Bl[n][c8], Bmat + (size_t)(cb + n) * K + kb + c8);
    }
    wait_async();
    __syncthreads();

    // A fragments: rows wave*32 + {0,16} + ln; K elems { half*8+0..7, 16+half*8+0..7 }
    Frag16 a[2];
#pragma unroll
    for (int rr = 0; rr < 2; ++rr) {
      const int arow = wave * 32 + rr * 16 + ln;
      a[rr].q[0] = *(const uint4*)(&Al[arow][half * 8]);
      a[rr].q[1] = *(const uint4*)(&Al[arow][16 + half * 8]);
    }
#pragma unroll
    for (int tn = 0; tn < 4; ++tn) {
      // B fragment: col n = tn*16+ln; K elems { half*8+0..7, 16+half*8+0..7 }
      Frag16 b;
      const int brow = tn * 16 + ln;
      b.q[0] = *(const uint4*)(&Bl[brow][half * 8]);
      b.q[1] = *(const uint4*)(&Bl[brow][16 + half * 8]);
#pragma unroll
      for (int rr = 0; rr < 2; ++rr)
        acc[rr][tn] = wmma_bf16(a[rr].v, b.v, acc[rr][tn]);
    }
  }

  // Bias + store. C layout: lane half -> rows half*8 + i, col = ln within tile.
#pragma unroll
  for (int rr = 0; rr < 2; ++rr)
#pragma unroll
    for (int tn = 0; tn < 4; ++tn) {
      const int col = cb + tn * 16 + ln;
      const float bv = bz[col];
#pragma unroll
      for (int i = 0; i < 8; ++i) {
        const int row = rowBase + wave * 32 + rr * 16 + half * 8 + i;
        const float v = acc[rr][tn][i] + bv;
        if (OUT_BF16)
          outH[(size_t)z * outStride + (size_t)row * N + col] = (bf16_t)v;
        else
          outF[(size_t)row * N + col] = v;
      }
    }
}

// ---------------------------------------------------------------------------
// Flash attention per (head, batch): S = Q K^T + mask, online softmax, O = P V.
// Grid: x = qRowBlock (L/128 = 8), y = head*32 + batch (256). 256 threads.
// Staging: Q via async loads, K via TDM tensor_load_to_lds (contiguous 8KB
// block -> 1D descriptor), V via manual transpose.
// ---------------------------------------------------------------------------
__global__ __launch_bounds__(256) void flash_attn_wmma(
    const bf16_t* __restrict__ qkv, const float* __restrict__ mask,
    bf16_t* __restrict__ ctx) {
  const int qb = blockIdx.x;        // 0..7
  const int hd = blockIdx.y >> 5;   // head 0..7
  const int b  = blockIdx.y & 31;   // batch 0..31

  const bf16_t* Qg = qkv + ((size_t)(0 * HH + hd) * BL + (size_t)b * LL) * DD;
  const bf16_t* Kg = qkv + ((size_t)(1 * HH + hd) * BL + (size_t)b * LL) * DD;
  const bf16_t* Vg = qkv + ((size_t)(2 * HH + hd) * BL + (size_t)b * LL) * DD;

  __shared__ bf16_t Ql[128][72];    // [qrow][d], padded
  __shared__ bf16_t Kl[64][64];     // [key][d], unpadded (TDM writes contiguous 8KB)
  __shared__ bf16_t Vt[64][72];     // [d][key] (transposed), padded
  __shared__ bf16_t Pl[8][16][72];  // per-wave P re-layout buffer [wave][row][j]

  const int tid  = threadIdx.x;
  const int wave = tid >> 5;
  const int lane = tid & 31;
  const int half = lane >> 4;
  const int ln   = lane & 15;
  const int qr0  = qb * 128;

  // Async-stage Q block once: 128 rows x 64 d = 1024 b128 chunks.
#pragma unroll
  for (int c = tid; c < 1024; c += 256) {
    int r  = c >> 3;
    int c8 = (c & 7) * 8;
    async_copy_b128(&Ql[r][c8], Qg + (size_t)(qr0 + r) * DD + c8);
  }
  wait_async();
  __syncthreads();

  // Q fragments for this wave's 16 rows, both d-chunks; kept in VGPRs all kernel.
  Frag16 aq[2];
  const int qrow = wave * 16 + ln;
#pragma unroll
  for (int kc = 0; kc < 2; ++kc) {
    aq[kc].q[0] = *(const uint4*)(&Ql[qrow][kc * 32 + half * 8]);
    aq[kc].q[1] = *(const uint4*)(&Ql[qrow][kc * 32 + 16 + half * 8]);
  }

  float run_m[8], run_l[8];
  v8f O[4];
#pragma unroll
  for (int i = 0; i < 8; ++i) { run_m[i] = -1e30f; run_l[i] = 0.f; }
#pragma unroll
  for (int t = 0; t < 4; ++t) O[t] = (v8f){0.f, 0.f, 0.f, 0.f, 0.f, 0.f, 0.f, 0.f};

  const int grow0 = qr0 + wave * 16 + half * 8;  // + i gives global q row

  for (int kt = 0; kt < LL / 64; ++kt) {
    __syncthreads();

    // --- K block via Tensor Data Mover: one contiguous 8KB region. ---
    if (wave == 0) {
      uint64_t ga = (uint64_t)(uintptr_t)(Kg + (size_t)kt * 64 * DD);
      uint32_t la = (uint32_t)(uintptr_t)(&Kl[0][0]);
      u32x4 g0;
      g0[0] = 1u;                                            // count=1, user mode
      g0[1] = la;                                            // lds_addr
      g0[2] = (uint32_t)ga;                                  // global_addr[31:0]
      g0[3] = (uint32_t)((ga >> 32) & 0x1FFFFFFu) | (2u << 30);  // addr[56:32] | type=2
      u32x8 g1;
      g1[0] = 3u << 16;              // workgroup_mask=0, data_size=3 (8B units)
      g1[1] = (1024u & 0xFFFFu) << 16;  // tensor_dim0 = 1024 units (low 16)
      g1[2] = 1u << 16;              // tensor_dim0 hi=0 | tensor_dim1 lo16 = 1
      g1[3] = 1024u << 16;           // tensor_dim1 hi=0 | tile_dim0 = 1024
      g1[4] = 0u;                    // tile_dim1 = 0 (1D), tile_dim2 = 0
      g1[5] = 1024u;                 // tensor_dim0_stride (low 32) = 1024 units
      g1[6] = 0u;
      g1[7] = 0u;
      asm volatile("tensor_load_to_lds %0, %1" :: "s"(g0), "s"(g1) : "memory");
    }

    // --- V block, transposed into Vt[d][key] (needs element scatter). ---
#pragma unroll
    for (int c = tid; c < 512; c += 256) {
      int r  = c >> 3;          // key
      int c8 = (c & 7) * 8;     // d base
      uint4 d = *(const uint4*)(Vg + (size_t)(kt * 64 + r) * DD + c8);
      const bf16_t* e = (const bf16_t*)&d;
#pragma unroll
      for (int j = 0; j < 8; ++j) Vt[c8 + j][r] = e[j];
    }

    if (wave == 0) __builtin_amdgcn_s_wait_tensorcnt(0);
    __syncthreads();

    // S = Q K^T for 4 key tiles of 16.
    v8f s[4];
#pragma unroll
    for (int tn = 0; tn < 4; ++tn) {
      v8f acc = (v8f){0.f, 0.f, 0.f, 0.f, 0.f, 0.f, 0.f, 0.f};
#pragma unroll
      for (int kc = 0; kc < 2; ++kc) {
        // B frag: col = key tn*16+ln; K(=d) elems = kc*32 + { half*8+0..7, 16+half*8+0..7 }
        Frag16 bk;
        const int krow = tn * 16 + ln;
        bk.q[0] = *(const uint4*)(&Kl[krow][kc * 32 + half * 8]);
        bk.q[1] = *(const uint4*)(&Kl[krow][kc * 32 + 16 + half * 8]);
        acc = wmma_bf16(aq[kc].v, bk.v, acc);
      }
      s[tn] = acc;
    }

    // Add mask (same for all h,b -> L2 resident).
#pragma unroll
    for (int tn = 0; tn < 4; ++tn) {
      const int kcol = kt * 64 + tn * 16 + ln;
#pragma unroll
      for (int i = 0; i < 8; ++i)
        s[tn][i] += mask[(size_t)(grow0 + i) * LL + kcol];
    }

    // Online softmax. Rows live in lane halves: reduce over 16 lanes of a half.
    float mb[8];
#pragma unroll
    for (int i = 0; i < 8; ++i) {
      float m = fmaxf(fmaxf(s[0][i], s[1][i]), fmaxf(s[2][i], s[3][i]));
#pragma unroll
      for (int off = 1; off < 16; off <<= 1)
        m = fmaxf(m, __shfl_xor(m, off, 32));
      mb[i] = m;
    }
    float scale[8];
#pragma unroll
    for (int i = 0; i < 8; ++i) {
      float nm = fmaxf(run_m[i], mb[i]);
      scale[i] = __expf(run_m[i] - nm);
      run_m[i] = nm;
    }
    float lb[8];
#pragma unroll
    for (int i = 0; i < 8; ++i) lb[i] = 0.f;
#pragma unroll
    for (int tn = 0; tn < 4; ++tn)
#pragma unroll
      for (int i = 0; i < 8; ++i) {
        float p = __expf(s[tn][i] - run_m[i]);
        s[tn][i] = p;
        lb[i] += p;
      }
#pragma unroll
    for (int i = 0; i < 8; ++i) {
#pragma unroll
      for (int off = 1; off < 16; off <<= 1)
        lb[i] += __shfl_xor(lb[i], off, 32);
      run_l[i] = run_l[i] * scale[i] + lb[i];
    }
#pragma unroll
    for (int tn = 0; tn < 4; ++tn)
#pragma unroll
      for (int i = 0; i < 8; ++i) O[tn][i] *= scale[i];

    // Re-layout P (C layout -> A layout) through this wave's private LDS region.
#pragma unroll
    for (int tn = 0; tn < 4; ++tn)
#pragma unroll
      for (int i = 0; i < 8; ++i)
        Pl[wave][half * 8 + i][tn * 16 + ln] = (bf16_t)s[tn][i];

    Frag16 ap[2];
#pragma unroll
    for (int jc = 0; jc < 2; ++jc) {
      ap[jc].q[0] = *(const uint4*)(&Pl[wave][ln][jc * 32 + half * 8]);
      ap[jc].q[1] = *(const uint4*)(&Pl[wave][ln][jc * 32 + 16 + half * 8]);
    }

    // O += P V for 4 d-tiles.
#pragma unroll
    for (int dn = 0; dn < 4; ++dn) {
#pragma unroll
      for (int jc = 0; jc < 2; ++jc) {
        Frag16 bv;
        const int drow = dn * 16 + ln;
        bv.q[0] = *(const uint4*)(&Vt[drow][jc * 32 + half * 8]);
        bv.q[1] = *(const uint4*)(&Vt[drow][jc * 32 + 16 + half * 8]);
        O[dn] = wmma_bf16(ap[jc].v, bv.v, O[dn]);
      }
    }
  }

  // Normalize and store ctx (BL x M), col = hd*64 + d.
  float inv[8];
#pragma unroll
  for (int i = 0; i < 8; ++i) inv[i] = 1.0f / run_l[i];
#pragma unroll
  for (int dn = 0; dn < 4; ++dn) {
    const int col = hd * DD + dn * 16 + ln;
#pragma unroll
    for (int i = 0; i < 8; ++i) {
      const int row = b * LL + qr0 + wave * 16 + half * 8 + i;
      ctx[(size_t)row * MM + col] = (bf16_t)(O[dn][i] * inv[i]);
    }
  }
}

// ---------------------------------------------------------------------------
// Host-side launch
// ---------------------------------------------------------------------------
extern "C" void kernel_launch(void* const* d_in, const int* in_sizes, int n_in,
                              void* d_out, int out_size, void* d_ws, size_t ws_size,
                              hipStream_t stream) {
  const float* inp     = (const float*)d_in[0];  // (BL, M)
  const float* mask    = (const float*)d_in[1];  // (1,1,L,L)
  const float* pre_w   = (const float*)d_in[2];  // (3,H,M,D)
  const float* pre_b   = (const float*)d_in[3];  // (3,H,1,D)
  const float* post_w  = (const float*)d_in[4];  // (M,M)
  const float* post_b  = (const float*)d_in[5];  // (M,)
  float* out = (float*)d_out;                    // (B,L,M)

  // Workspace layout (bf16), all 16B aligned:
  char* ws = (char*)d_ws;
  size_t off = 0;
  bf16_t* inp_bf    = (bf16_t*)(ws + off); off += (size_t)BL * MM * 2;          // 32 MB
  bf16_t* prew_t    = (bf16_t*)(ws + off); off += (size_t)3 * HH * MM * DD * 2; // 1.5 MB (z, D, M)
  bf16_t* postw_t   = (bf16_t*)(ws + off); off += (size_t)MM * MM * 2;          // 0.5 MB (N, K)
  bf16_t* qkv_bf    = (bf16_t*)(ws + off); off += (size_t)3 * HH * BL * DD * 2; // 96 MB
  bf16_t* ctx_bf    = (bf16_t*)(ws + off); off += (size_t)BL * MM * 2;          // 32 MB
  (void)off; (void)ws_size; (void)in_sizes; (void)n_in; (void)out_size;

  // 1) Convert activations (linear) and weights (transposed) to bf16.
  {
    int n8 = (BL * MM) / 8;  // 2097152
    cvt_f32_to_bf16<<<(n8 + 255) / 256, 256, 0, stream>>>(inp, inp_bf, n8);
  }
  {
    // pre_w: per z=(t,h), (512 x 64) -> (64 x 512): 64*(512/8)=4096 threads per z
    dim3 grid(16, 1, 3 * HH);
    transpose_cvt_bf16<<<grid, 256, 0, stream>>>(pre_w, prew_t, MM, DD);
  }
  {
    // post_w: (512 x 512) -> (512 x 512)^T: 512*64 = 32768 threads
    dim3 grid(128, 1, 1);
    transpose_cvt_bf16<<<grid, 256, 0, stream>>>(post_w, postw_t, MM, MM);
  }

  // 2) QKV projection: 24 GEMMs (BL x 512) * (512 x 64) + bias -> qkv_bf (z, BL, 64)
  {
    dim3 grid(BL / 256, 1, 3 * HH);
    gemm_bias_wmma<true><<<grid, 256, 0, stream>>>(
        inp_bf, prew_t, pre_b, nullptr, qkv_bf,
        /*K=*/MM, /*N=*/DD,
        /*Bstride=*/(long)MM * DD, /*biasStride=*/DD,
        /*outStride=*/(long)BL * DD);
  }

  // 3) Flash attention -> ctx_bf (BL, M)
  {
    dim3 grid(LL / 128, HH * BB);
    flash_attn_wmma<<<grid, 256, 0, stream>>>(qkv_bf, mask, ctx_bf);
  }

  // 4) Output projection: (BL x 512) * (512 x 512) + bias -> f32 out
  {
    dim3 grid(BL / 256, MM / 64, 1);
    gemm_bias_wmma<false><<<grid, 256, 0, stream>>>(
        ctx_bf, postw_t, post_b, out, nullptr,
        /*K=*/MM, /*N=*/MM,
        /*Bstride=*/0, /*biasStride=*/0, /*outStride=*/0);
  }
}
